// TSModel_32770600468945
// MI455X (gfx1250) — matrix operand order repaired and checked by reading, not compiled, not run on hardware
//
#include <hip/hip_runtime.h>

// Problem dims (fixed by the reference)
#define Bsz 16      // batch (== WMMA M)
#define Tn  64      // timesteps
#define Hn  256     // hidden
#define Ln  64      // layers
#define Kn  512     // 2H   (GEMM K)
#define Nn  1024    // 4H   (GEMM N)
#define KSTEPS 16   // Kn / 32 (bf16 WMMA K=32)
#define NTHREADS 512 // 16 waves

typedef __attribute__((ext_vector_type(16))) __bf16 bf16x16;
typedef __attribute__((ext_vector_type(8)))  float  floatx8;

static __device__ __forceinline__ unsigned short f32_to_bf16_rn(float f) {
  union { float f; unsigned int u; } v; v.f = f;
  unsigned int u = v.u;
  u += 0x7fffu + ((u >> 16) & 1u);   // round-to-nearest-even
  return (unsigned short)(u >> 16);
}
static __device__ __forceinline__ float bf16_to_f32(unsigned short h) {
  union { unsigned int u; float f; } v; v.u = ((unsigned int)h) << 16;
  return v.f;
}
// Fast gate nonlinearities: v_rcp_f32 (~1 ulp) instead of IEEE divide chains,
// v_tanh_f32 if the toolchain exposes it.
static __device__ __forceinline__ float sigm(float x) {
  return __builtin_amdgcn_rcpf(1.0f + __expf(-x));
}
static __device__ __forceinline__ float tanh_fast(float x) {
#if __has_builtin(__builtin_amdgcn_tanhf)
  return __builtin_amdgcn_tanhf(x);
#else
  float e = __expf(-2.0f * x);
  return (1.0f - e) * __builtin_amdgcn_rcpf(1.0f + e);
#endif
}

// ---------------------------------------------------------------------------
// Pre-pack W (512x1024 f32, row-major k*Nn+n) into hi/lo bf16 in this file's
// WMMA B-operand (32x16 bf16) lane layout (contiguous k, matching the A staging):
//   packed idx = ((tile*KSTEPS + ks)*32 + lane)*16 + e
//   lane = half*16 + n_in_tile ; k = ks*32 + e + 16*half
// Also zeroes the progress flags for the persistent kernel.
// ---------------------------------------------------------------------------
__global__ void pack_weights(const float* __restrict__ W,
                             unsigned short* __restrict__ Wh,
                             unsigned short* __restrict__ Wl,
                             int* __restrict__ prog) {
  int tid = blockIdx.x * blockDim.x + threadIdx.x;
  if (blockIdx.x == 0 && threadIdx.x < Ln) prog[threadIdx.x] = 0;
  if (tid >= Kn * Nn) return;
  int e    = tid & 15;
  int lane = (tid >> 4) & 31;
  int ks   = (tid >> 9) & 15;
  int tile = tid >> 13;
  int hlf  = lane >> 4;
  int n    = tile * 16 + (lane & 15);
  int k    = ks * 32 + e + 16 * hlf;
  float w  = W[(size_t)k * Nn + n];
  unsigned short hb = f32_to_bf16_rn(w);
  Wh[tid] = hb;
  Wl[tid] = f32_to_bf16_rn(w - bf16_to_f32(hb));   // residual -> ~16 mantissa bits
}

// ---------------------------------------------------------------------------
// Persistent systolic LSTM: one block (16 waves, 512 threads) per layer.
// Block l loops t = 0..63; waits on prog[l-1] >= t+1 (input h ready) and
// prog[l+1] >= t-1 (parity slot of the h double-buffer free to overwrite).
//
// Wave tile map: wave w owns N-tile g*16 + w for each gate g in {i,j,f,o},
// i.e. h-elements e in [16w, 16w+16) for ALL four gates -> the LSTM
// nonlinearity is wave/register local. Cell state c stays in VGPRs (same
// layout as the WMMA accumulator) for all 64 timesteps.
// ---------------------------------------------------------------------------
__global__ void __launch_bounds__(NTHREADS, 1)
lstm_persistent(const float* __restrict__ x,            // (B,T,H)
                const float* __restrict__ init_state,   // (L,2,B,H)
                const float* __restrict__ bias,         // (4H)
                const unsigned short* __restrict__ Wh,  // packed hi
                const unsigned short* __restrict__ Wl,  // packed lo
                float* __restrict__ hbuf,               // (L,2,B,H) double buffer
                int* prog,                              // (L) progress flags
                float* __restrict__ out) {              // (B,T,H)
  __shared__ alignas(32) unsigned short Ah[Bsz * Kn];   // 16 KB: A hi, WMMA layout
  __shared__ alignas(32) unsigned short Al[Bsz * Kn];   // 16 KB: A lo
  __shared__ float hloc[Bsz * Hn];                      // 16 KB: h(l, t-1)

  const int l    = blockIdx.x;
  const int tid  = threadIdx.x;
  const int lane = tid & 31;
  const int wv   = tid >> 5;        // 0..15
  const int nsub = lane & 15;
  const int hlf  = lane >> 4;

  // c state: register-resident, C/D-matrix layout (m = r + 8*hlf, e = 16*wv + nsub)
  floatx8 cacc;
  #pragma unroll
  for (int r = 0; r < 8; ++r) {
    int m = r + 8 * hlf;
    int e = wv * 16 + nsub;
    cacc[r] = init_state[((size_t)(l * 2 + 0) * Bsz + m) * Hn + e];
  }
  // h0 -> LDS
  for (int idx = tid; idx < Bsz * Hn; idx += NTHREADS)
    hloc[idx] = init_state[(size_t)(l * 2 + 1) * Bsz * Hn + idx];
  __syncthreads();

  // per-gate bias (same for every row m) and per-wave B base pointers
  float bv[4];
  const unsigned short* pWh[4];
  const unsigned short* pWl[4];
  #pragma unroll
  for (int g = 0; g < 4; ++g) {
    int tile = g * 16 + wv;
    bv[g] = bias[tile * 16 + nsub];
    int boff = tile * (KSTEPS * 32 * 16) + lane * 16;   // elements
    pWh[g] = Wh + boff;
    pWl[g] = Wl + boff;
  }

  for (int t = 0; t < Tn; ++t) {
    // ---- 1. wait on pipeline dependencies --------------------------------
    if (tid == 0) {
      if (l > 0)
        while (__atomic_load_n(&prog[l - 1], __ATOMIC_ACQUIRE) < t + 1)
          __builtin_amdgcn_s_sleep(1);
      if (l < Ln - 1 && t >= 2)
        while (__atomic_load_n(&prog[l + 1], __ATOMIC_ACQUIRE) < t - 1)
          __builtin_amdgcn_s_sleep(1);
    }
    __syncthreads();

    // ---- 2. stage A = [inp | h_prev] as hi/lo bf16 in WMMA A layout ------
    //   A 16x32 tile: lane = half*16 + m, e = kr&15, half=kr>>4
    const float* hin = hbuf + (size_t)((l > 0 ? l - 1 : 0) * 2 + (t & 1)) * Bsz * Hn;
    for (int idx = tid; idx < Bsz * Kn; idx += NTHREADS) {
      int m = idx & 15;
      int k = idx >> 4;
      float v;
      if (k < Hn) v = l ? hin[m * Hn + k] : x[((size_t)m * Tn + t) * Hn + k];
      else        v = hloc[m * Hn + (k - Hn)];
      int ks2 = k >> 5;
      int kr  = k & 31;
      int hh  = kr >> 4;
      int e   = kr & 15;
      int off = (ks2 * 32 + (hh * 16 + m)) * 16 + e;
      unsigned short hb = f32_to_bf16_rn(v);
      Ah[off] = hb;
      Al[off] = f32_to_bf16_rn(v - bf16_to_f32(hb));
    }
    __syncthreads();

    // ---- 3. GEMM: split-bf16 (Ah*Wh + Ah*Wl + Al*Wh), fp32 accumulate ----
    floatx8 acc[4];
    #pragma unroll
    for (int g = 0; g < 4; ++g) {
      floatx8 z;
      #pragma unroll
      for (int r = 0; r < 8; ++r) z[r] = bv[g];
      acc[g] = z;
    }
    #pragma unroll 1   // keep the scheduling window small: no spills
    for (int ks2 = 0; ks2 < KSTEPS; ++ks2) {
      bf16x16 a_hi = *(const bf16x16*)&Ah[(ks2 * 32 + lane) * 16];
      bf16x16 a_lo = *(const bf16x16*)&Al[(ks2 * 32 + lane) * 16];
      int koff = ks2 * 512;                    // elements per K-step per tile
      #pragma unroll
      for (int g = 0; g < 4; ++g) {
        bf16x16 b_hi = *(const bf16x16*)(pWh[g] + koff);
        bf16x16 b_lo = *(const bf16x16*)(pWl[g] + koff);
        acc[g] = __builtin_amdgcn_wmma_f32_16x16x32_bf16(
                     false, a_hi, false, b_hi, (short)0, acc[g], false, false);
        acc[g] = __builtin_amdgcn_wmma_f32_16x16x32_bf16(
                     false, a_hi, false, b_lo, (short)0, acc[g], false, false);
        acc[g] = __builtin_amdgcn_wmma_f32_16x16x32_bf16(
                     false, a_lo, false, b_hi, (short)0, acc[g], false, false);
      }
    }

    // ---- 4. gates + state update (all register local) --------------------
    float* hout = hbuf + (size_t)(l * 2 + (t & 1)) * Bsz * Hn;
    #pragma unroll
    for (int r = 0; r < 8; ++r) {
      float zi = acc[0][r];
      float zj = acc[1][r];
      float zf = acc[2][r];
      float zo = acc[3][r];
      float nc = sigm(zf + 1.0f) * cacc[r] + sigm(zi) * tanh_fast(zj);
      cacc[r] = nc;
      float nh = sigm(zo) * tanh_fast(nc);
      int m = r + 8 * hlf;
      int e = wv * 16 + nsub;
      hloc[m * Hn + e] = nh;                              // own next-step input
      if (l < Ln - 1) hout[m * Hn + e] = nh;              // feed layer l+1
      else            out[((size_t)m * Tn + t) * Hn + e] = nh;
    }
    __threadfence();     // every thread: drain its global h stores
    __syncthreads();
    if (tid == 0)
      __atomic_store_n(&prog[l], t + 1, __ATOMIC_RELEASE);  // publish cell done
  }
}

// ---------------------------------------------------------------------------
// d_in: 0=x (B,T,H) f32 | 1=init_state (L,2,B,H) f32 | 2=W (2H,4H) f32 | 3=b (4H)
// ws layout: Wh 1MB | Wl 1MB | hbuf 2MB | prog 256B   (~4.2 MB total)
// ---------------------------------------------------------------------------
extern "C" void kernel_launch(void* const* d_in, const int* in_sizes, int n_in,
                              void* d_out, int out_size, void* d_ws, size_t ws_size,
                              hipStream_t stream) {
  (void)in_sizes; (void)n_in; (void)out_size; (void)ws_size;
  const float* x  = (const float*)d_in[0];
  const float* st = (const float*)d_in[1];
  const float* W  = (const float*)d_in[2];
  const float* b  = (const float*)d_in[3];
  char* ws = (char*)d_ws;
  unsigned short* Wh  = (unsigned short*)(ws);
  unsigned short* Wl  = (unsigned short*)(ws + (size_t)Kn * Nn * 2);
  float*          hb  = (float*)        (ws + (size_t)Kn * Nn * 4);
  int*            prg = (int*)          (ws + (size_t)Kn * Nn * 4
                                            + (size_t)Ln * 2 * Bsz * Hn * 4);
  pack_weights<<<(Kn * Nn + 255) / 256, 256, 0, stream>>>(W, Wh, Wl, prg);
  lstm_persistent<<<Ln, NTHREADS, 0, stream>>>(x, st, b, Wh, Wl, hb, prg,
                                               (float*)d_out);
}
